// SnakeNet_59072980189811
// MI455X (gfx1250) — matrix operand order repaired and checked by reading, not compiled, not run on hardware
//
#include <hip/hip_runtime.h>
#include <hip/hip_bf16.h>

// ---------------------------------------------------------------------------
// SnakeNet forward for MI455X (gfx1250, wave32, WMMA).
// All GEMM-shaped math uses v_wmma_f32_16x16x32_f16 (f16 in, f32 accum).
// N-blocked WMMA: one wave gathers each A fragment once and issues NBLK
// back-to-back WMMAs against NBLK pre-packed B fragments.
// Conv biases before batchnorm are dropped (BN subtracts the mean).
// Backward snake runs in original time order: only the depthwise-conv
// neighbor offset, the scan direction and the output channel block differ.
// Boundary handling: clamped-address loads + value select (no divergence).
// Selective scan stages B/C in 64-step LDS chunks (2 barriers / 64 steps).
// Assumes ws_size >= ~224 MB.
// ---------------------------------------------------------------------------

typedef __attribute__((ext_vector_type(16))) _Float16 v16h;
typedef __attribute__((ext_vector_type(8)))  _Float16 v8h;
typedef __attribute__((ext_vector_type(8)))  float    v8f;

#define BATCH 8
#define LFULL 16384
#define L1    8192
#define L2    4096
#define DIM   64
#define DIN   256   // D_INNER
#define NPOS  (BATCH * L2)   // 32768 GEMM rows
#define SCHUNK 64            // scan LDS staging chunk

__device__ __forceinline__ float load_clamped(const float* __restrict__ p, int idx, int lim)
{
    int ic = idx < 0 ? 0 : (idx >= lim ? lim - 1 : idx);
    float v = p[ic];                 // always in-bounds: plain pipelined load
    return (idx == ic) ? v : 0.f;    // v_cndmask, no exec-mask churn
}

// ---------------------------------------------------------------------------
// Weight packing into the hardware B-fragment layout.
// Packed index = ((nt*Ktiles + kt)*32 + lane)*16 + e
// B layout: n = nt*16 + (lane&15); k = kt*32 + (e<8 ? e : e+8) + (lane>=16 ? 8 : 0)
// mode 0: B[k][n] = src[n*K + k]                       (dense W stored (N,K))
// mode 1: dec2 conv-transpose: k=(ci*16+j), B[k][n] = w[(ci*128+n)*32 + (31-2j-odd)]
// ---------------------------------------------------------------------------
__global__ void pack_b_kernel(const float* __restrict__ src, _Float16* __restrict__ Bp,
                              int K, int Ktiles, int Ntiles, int Nreal, int mode, int odd)
{
    int idx = blockIdx.x * blockDim.x + threadIdx.x;
    int total = Ktiles * Ntiles * 512;
    if (idx >= total) return;
    int e    = idx & 15;
    int lane = (idx >> 4) & 31;
    int tile = idx >> 9;
    int kt   = tile % Ktiles;
    int nt   = tile / Ktiles;
    int k = kt * 32 + (e < 8 ? e : e + 8) + ((lane >> 4) << 3);
    int n = nt * 16 + (lane & 15);
    float v = 0.f;
    if (n < Nreal && k < K) {
        if (mode == 0) {
            v = src[(size_t)n * K + k];
        } else {
            int ci = k >> 4, j = k & 15;
            v = src[((size_t)ci * 128 + n) * 32 + (31 - 2 * j - odd)];
        }
    }
    Bp[idx] = (_Float16)v;
}

// ---------------------------------------------------------------------------
// enc1: y1[b,c,t] = sum_k (x[b, 2t+k-7]+1) * w[c,k]   (bias absorbed by BN)
// ---------------------------------------------------------------------------
__global__ void enc1_kernel(const float* __restrict__ x, const float* __restrict__ w,
                            float* __restrict__ y1)
{
    size_t idx = (size_t)blockIdx.x * blockDim.x + threadIdx.x;
    if (idx >= (size_t)BATCH * 128 * L1) return;
    int t = idx % L1;
    int c = (idx / L1) % 128;
    int b = idx / ((size_t)L1 * 128);
    const float* xb = x + (size_t)b * LFULL;
    float acc = 0.f;
    #pragma unroll
    for (int k = 0; k < 16; ++k) {
        int p = 2 * t + k - 7;
        int pc = p < 0 ? 0 : (p >= LFULL ? LFULL - 1 : p);
        float xv = xb[pc];
        xv = (p == pc) ? (xv + 1.f) : 0.f;
        acc += xv * w[c * 16 + k];
    }
    y1[idx] = acc;
}

// ---------------------------------------------------------------------------
// Batchnorm stats: one block per channel. stats[2c]=mean, stats[2c+1]=rsqrt(var+eps)
// ---------------------------------------------------------------------------
__global__ void bn_stats_kernel(const float* __restrict__ src, float* __restrict__ stats,
                                int C, int Lc)
{
    int c = blockIdx.x, tid = threadIdx.x;
    float s = 0.f, s2 = 0.f;
    size_t per_b = (size_t)C * Lc;
    for (int b = 0; b < BATCH; ++b) {
        const float* p = src + (size_t)b * per_b + (size_t)c * Lc;
        for (int i = tid; i < Lc; i += 256) { float v = p[i]; s += v; s2 += v * v; }
    }
    __shared__ float sh[512];
    sh[tid] = s; sh[256 + tid] = s2;
    __syncthreads();
    for (int w = 128; w > 0; w >>= 1) {
        if (tid < w) { sh[tid] += sh[tid + w]; sh[256 + tid] += sh[256 + tid + w]; }
        __syncthreads();
    }
    if (tid == 0) {
        float n = (float)BATCH * (float)Lc;
        float m = sh[0] / n;
        float var = sh[256] / n - m * m;
        stats[c * 2]     = m;
        stats[c * 2 + 1] = rsqrtf(var + 1e-5f);
    }
}

__global__ void bn_relu_kernel(const float* __restrict__ src, float* __restrict__ dst,
                               const float* __restrict__ stats,
                               const float* __restrict__ g, const float* __restrict__ bb,
                               int C, int Lc)
{
    size_t idx = (size_t)blockIdx.x * blockDim.x + threadIdx.x;
    size_t total = (size_t)BATCH * C * Lc;
    if (idx >= total) return;
    int c = (int)((idx / Lc) % C);
    float v = (src[idx] - stats[c * 2]) * stats[c * 2 + 1] * g[c] + bb[c];
    dst[idx] = v > 0.f ? v : 0.f;
}

// ---------------------------------------------------------------------------
// enc2 as im2col WMMA GEMM: M=B*L2, N=64, K=128*32. One wave computes a
// 16x64 output strip (4 N-tiles) so the A gather is done once per K-tile.
// A[m=(b,t)][k=ci*32+kk] = x1[b,ci, 2t+kk-15]
// ---------------------------------------------------------------------------
__global__ void enc2_wmma_kernel(const float* __restrict__ x1, const _Float16* __restrict__ Bp,
                                 float* __restrict__ x2raw)
{
    const int NB = 4, Kt = 128;
    int mt = blockIdx.x, lane = threadIdx.x;
    int m = mt * 16 + (lane & 15);
    int b = m >> 12, t = m & 4095;
    int off = (lane >> 4) * 8;
    v8f c[NB];
    #pragma unroll
    for (int w = 0; w < NB; ++w) c[w] = (v8f){};
    for (int ci = 0; ci < Kt; ++ci) {
        const float* src = x1 + ((size_t)b * 128 + ci) * L1;
        int p0 = 2 * t - 15 + off;
        v16h a;
        #pragma unroll
        for (int e = 0; e < 16; ++e) {
            int p = p0 + (e < 8 ? e : 8 + e);  // kk = off + {0..7, 16..23}
            a[e] = (_Float16)load_clamped(src, p, L1);
        }
        #pragma unroll
        for (int w = 0; w < NB; ++w) {
            v16h bf = *(const v16h*)(Bp + (((size_t)w * Kt + ci) * 32 + lane) * 16);
            c[w] = __builtin_amdgcn_wmma_f32_16x16x32_f16(false, a, false, bf, (short)0, c[w], false, false);
        }
    }
    int mb = mt * 16 + off;
    #pragma unroll
    for (int w = 0; w < NB; ++w) {
        int n = w * 16 + (lane & 15);
        #pragma unroll
        for (int r = 0; r < 8; ++r) {
            int mm = mb + r, bb = mm >> 12, tt = mm & 4095;
            x2raw[((size_t)bb * 64 + n) * L2 + tt] = c[w][r];
        }
    }
}

// ---------------------------------------------------------------------------
// Generic N-blocked WMMA GEMM: A row-major f16 (lda mult of 8), B pre-packed.
// Wave computes 16 x (16*NBLK); A fragment loaded once per K-tile.
// out = f32 (outF) or f16 (outH), row-major with ldo (covers padded N).
// ---------------------------------------------------------------------------
template<int NBLK>
__global__ void gemm_wmma_kernel(const _Float16* __restrict__ A, int lda,
                                 const _Float16* __restrict__ Bp,
                                 float* __restrict__ outF, _Float16* __restrict__ outH,
                                 int ldo, int Ktiles)
{
    int mt = blockIdx.x, ntb = blockIdx.y, lane = threadIdx.x;
    int mrow = mt * 16 + (lane & 15);
    int off  = (lane >> 4) * 8;
    const _Float16* arow = A + (size_t)mrow * lda;
    v8f c[NBLK];
    #pragma unroll
    for (int w = 0; w < NBLK; ++w) c[w] = (v8f){};
    for (int kt = 0; kt < Ktiles; ++kt) {
        int kb = kt * 32;
        v8h a0 = *(const v8h*)(arow + kb + off);
        v8h a1 = *(const v8h*)(arow + kb + off + 16);
        v16h a;
        #pragma unroll
        for (int e = 0; e < 8; ++e) { a[e] = a0[e]; a[e + 8] = a1[e]; }
        #pragma unroll
        for (int w = 0; w < NBLK; ++w) {
            int nt = ntb * NBLK + w;
            v16h bf = *(const v16h*)(Bp + (((size_t)nt * Ktiles + kt) * 32 + lane) * 16);
            c[w] = __builtin_amdgcn_wmma_f32_16x16x32_f16(false, a, false, bf, (short)0, c[w], false, false);
        }
    }
    int mb = mt * 16 + off;
    #pragma unroll
    for (int w = 0; w < NBLK; ++w) {
        int n = (ntb * NBLK + w) * 16 + (lane & 15);
        #pragma unroll
        for (int r = 0; r < 8; ++r) {
            size_t o = (size_t)(mb + r) * ldo + n;
            if (outF) outF[o] = c[w][r]; else outH[o] = (_Float16)c[w][r];
        }
    }
}

// ---------------------------------------------------------------------------
// x3[b,t,d] = x2[b,d,t]
// ---------------------------------------------------------------------------
__global__ void transpose_kernel(const float* __restrict__ x2, float* __restrict__ x3)
{
    size_t idx = (size_t)blockIdx.x * blockDim.x + threadIdx.x;
    if (idx >= (size_t)NPOS * 64) return;
    int d = idx & 63;
    size_t m = idx >> 6;
    int t = (int)(m & 4095);
    int b = (int)(m >> 12);
    x3[idx] = x2[((size_t)b * 64 + d) * L2 + t];
}

__global__ void rmsnorm_kernel(const float* __restrict__ x3, const float* __restrict__ nw,
                               _Float16* __restrict__ hn)
{
    int m = blockIdx.x * blockDim.x + threadIdx.x;
    if (m >= NPOS) return;
    const float* p = x3 + (size_t)m * 64;
    float s = 0.f;
    for (int d = 0; d < 64; ++d) { float v = p[d]; s += v * v; }
    float r = rsqrtf(s / 64.f + 1e-5f);
    for (int d = 0; d < 64; ++d) hn[(size_t)m * 64 + d] = (_Float16)(p[d] * r * nw[d]);
}

// ---------------------------------------------------------------------------
// Depthwise causal conv (k=2) + SiLU.  prev index = t - dir (dir=+1 fwd, -1 bwd)
// ---------------------------------------------------------------------------
__global__ void conv_silu_kernel(const _Float16* __restrict__ xz, const float* __restrict__ cw,
                                 const float* __restrict__ cb, _Float16* __restrict__ u_h, int dir)
{
    size_t idx = (size_t)blockIdx.x * blockDim.x + threadIdx.x;
    if (idx >= (size_t)NPOS * DIN) return;
    int d = idx & 255;
    size_t m = idx >> 8;
    int t = (int)(m & 4095);
    size_t b = m >> 12;
    float x0 = (float)xz[m * 512 + d];
    int tp = t - dir;
    int tpc = tp < 0 ? 0 : (tp >= L2 ? L2 - 1 : tp);
    float xp = (float)xz[(((b << 12) + tpc) * 512) + d];
    xp = (tp == tpc) ? xp : 0.f;
    float v = xp * cw[d * 2] + x0 * cw[d * 2 + 1] + cb[d];
    u_h[idx] = (_Float16)(v / (1.f + __expf(-v)));
}

// dt = softplus(x_dbl[:, :4] @ dt_proj_w^T + dt_proj_b)
__global__ void dt_kernel(const float* __restrict__ xdbl, const float* __restrict__ dpw,
                          const float* __restrict__ dpb, float* __restrict__ dt)
{
    size_t idx = (size_t)blockIdx.x * blockDim.x + threadIdx.x;
    if (idx >= (size_t)NPOS * DIN) return;
    int d = idx & 255;
    size_t m = idx >> 8;
    const float* xr = xdbl + m * 32;
    float acc = dpb[d];
    #pragma unroll
    for (int r = 0; r < 4; ++r) acc += xr[r] * dpw[d * 4 + r];
    dt[idx] = (acc > 20.f) ? acc : log1pf(__expf(acc));
}

// ---------------------------------------------------------------------------
// Fused selective scan: one block per batch, thread d = inner channel.
// B/C staged in LDS in 64-step chunks: 2 barriers per 64 steps.
// h[n] <- h[n]*exp(dt*A[n]) + dt*u*B[n]; y = <h,C>; out = (y+u*D)*silu(z).
// ---------------------------------------------------------------------------
__global__ void scan_kernel(const _Float16* __restrict__ u_h, const float* __restrict__ dt,
                            const float* __restrict__ xdbl, const _Float16* __restrict__ xz,
                            const float* __restrict__ A_log, const float* __restrict__ Dp,
                            _Float16* __restrict__ y_h, int dir)
{
    int b = blockIdx.x, d = threadIdx.x;
    float A[8], h[8];
    #pragma unroll
    for (int n = 0; n < 8; ++n) { A[n] = -__expf(A_log[d * 8 + n]); h[n] = 0.f; }
    float Dd = Dp[d];
    __shared__ float sbc[SCHUNK][16];
    for (int cc = 0; cc < L2 / SCHUNK; ++cc) {
        int chunk = (dir > 0) ? cc : (L2 / SCHUNK - 1 - cc);
        int t0 = chunk * SCHUNK;
        // stage B/C for SCHUNK steps: 64*16 floats, 4 per thread, coalesced
        for (int q = d; q < SCHUNK * 16; q += 256) {
            int tt = q >> 4, e = q & 15;
            sbc[tt][e] = xdbl[(((size_t)b << 12) + t0 + tt) * 32 + 4 + e];
        }
        __syncthreads();
        for (int k = 0; k < SCHUNK; ++k) {
            int tl = (dir > 0) ? k : (SCHUNK - 1 - k);
            size_t m = ((size_t)b << 12) + t0 + tl;
            float dtv = dt[m * 256 + d];
            float uv  = (float)u_h[m * 256 + d];
            float zv  = (float)xz[m * 512 + 256 + d];
            float du  = dtv * uv;
            float acc = 0.f;
            #pragma unroll
            for (int n = 0; n < 8; ++n) {
                h[n] = h[n] * __expf(dtv * A[n]) + du * sbc[tl][n];
                acc += h[n] * sbc[tl][8 + n];
            }
            float yv = (acc + uv * Dd) * (zv / (1.f + __expf(-zv)));
            y_h[m * 256 + d] = (_Float16)yv;
        }
        __syncthreads();
    }
}

// ct[m,choff+d] = out_proj_result[m,d] + x3[m,d]   (residual add)
__global__ void add_ct_kernel(const float* __restrict__ tmp, const float* __restrict__ x3,
                              float* __restrict__ ct, int choff)
{
    size_t idx = (size_t)blockIdx.x * blockDim.x + threadIdx.x;
    if (idx >= (size_t)NPOS * 64) return;
    int d = idx & 63;
    size_t m = idx >> 6;
    ct[m * 128 + choff + d] = tmp[idx] + x3[idx];
}

// ---------------------------------------------------------------------------
// dec2 conv-transpose as im2col WMMA GEMM (even/odd t parity via blockIdx.z).
// One wave computes a 16x128 strip (all 8 N-tiles): A gathered once per K-tile.
// K = 192*16; A[m=(b,te)][k=ci*16+j] = d1[b,ci, te+j-8+parity]
// d1: ci<128 -> ct[b, s, ci]; ci>=128 -> x2[b, ci-128, s]
// ---------------------------------------------------------------------------
__global__ void dec2_wmma_kernel(const float* __restrict__ ct, const float* __restrict__ x2,
                                 const _Float16* __restrict__ BpE, const _Float16* __restrict__ BpO,
                                 float* __restrict__ d2raw)
{
    const int NB = 8, Kt = 96;
    int mt = blockIdx.x, parity = blockIdx.z, lane = threadIdx.x;
    const _Float16* Bp = parity ? BpO : BpE;
    int m = mt * 16 + (lane & 15);
    int b = m >> 12, te = m & 4095;
    int off = (lane >> 4) * 8;
    v8f c[NB];
    #pragma unroll
    for (int w = 0; w < NB; ++w) c[w] = (v8f){};
    for (int kt = 0; kt < Kt; ++kt) {
        v16h a;
        #pragma unroll
        for (int e = 0; e < 16; ++e) {
            int klocal = off + (e < 8 ? e : 8 + e);
            int ci = kt * 2 + (klocal >> 4);
            int j  = klocal & 15;
            int s  = te + j - 8 + parity;
            int sc = s < 0 ? 0 : (s >= L2 ? L2 - 1 : s);
            float v = (ci < 128) ? ct[((size_t)b * L2 + sc) * 128 + ci]
                                 : x2[((size_t)b * 64 + (ci - 128)) * L2 + sc];
            v = (s == sc) ? v : 0.f;
            a[e] = (_Float16)v;
        }
        #pragma unroll
        for (int w = 0; w < NB; ++w) {
            v16h bf = *(const v16h*)(Bp + (((size_t)w * Kt + kt) * 32 + lane) * 16);
            c[w] = __builtin_amdgcn_wmma_f32_16x16x32_f16(false, a, false, bf, (short)0, c[w], false, false);
        }
    }
    int mb = mt * 16 + off;
    #pragma unroll
    for (int w = 0; w < NB; ++w) {
        int n = w * 16 + (lane & 15);
        #pragma unroll
        for (int r = 0; r < 8; ++r) {
            int mm = mb + r, bb = mm >> 12, tte = mm & 4095;
            d2raw[((size_t)bb * 128 + n) * L1 + (2 * tte + parity)] = c[w][r];
        }
    }
}

// ---------------------------------------------------------------------------
// dec3: final conv-transpose to 1 channel (+bias, -1).
// t even: s=t/2+j-4, w[ci,15-2j]; t odd: s=(t-1)/2+j-3, w[ci,14-2j], j<8.
// channels: 0..127 = d2 (post-BN/relu), 128..255 = x1.
// ---------------------------------------------------------------------------
__global__ void dec3_kernel(const float* __restrict__ d2, const float* __restrict__ x1,
                            const float* __restrict__ w3, const float* __restrict__ b3,
                            float* __restrict__ out)
{
    size_t idx = (size_t)blockIdx.x * blockDim.x + threadIdx.x;
    if (idx >= (size_t)BATCH * LFULL) return;
    int t = (int)(idx & (LFULL - 1));
    int b = (int)(idx >> 14);
    int parity = t & 1;
    int s0 = parity ? ((t - 1) / 2 - 3) : (t / 2 - 4);
    float acc = b3[0];
    for (int ci = 0; ci < 256; ++ci) {
        const float* src = (ci < 128) ? d2 + ((size_t)b * 128 + ci) * L1
                                      : x1 + ((size_t)b * 128 + (ci - 128)) * L1;
        const float* wr = w3 + ci * 16;
        #pragma unroll
        for (int j = 0; j < 8; ++j) {
            int s = s0 + j;
            acc += load_clamped(src, s, L1) * wr[parity ? (14 - 2 * j) : (15 - 2 * j)];
        }
    }
    out[idx] = acc - 1.f;
}

// ---------------------------------------------------------------------------
// Host orchestration
// ---------------------------------------------------------------------------
static inline dim3 g1(size_t n, int bs = 256) { return dim3((unsigned)((n + bs - 1) / bs)); }

extern "C" void kernel_launch(void* const* d_in, const int* in_sizes, int n_in,
                              void* d_out, int out_size, void* d_ws, size_t ws_size,
                              hipStream_t stream) {
    (void)in_sizes; (void)n_in; (void)out_size; (void)ws_size;
    // Input order: x, enc1_w, enc1_b, bn1_g, bn1_b, enc2_w, enc2_b, bn2_g, bn2_b,
    // dec2_w, dec2_b, bnd2_g, bnd2_b, dec3_w, dec3_b,
    // snake{in_proj_w,conv_w,conv_b,x_proj_w,dt_proj_w,dt_proj_b,A_log,D,out_proj_w,norm_w},
    // bsnake{...same 10...}
    const float* x       = (const float*)d_in[0];
    const float* enc1_w  = (const float*)d_in[1];
    const float* bn1_g   = (const float*)d_in[3];
    const float* bn1_b   = (const float*)d_in[4];
    const float* enc2_w  = (const float*)d_in[5];
    const float* bn2_g   = (const float*)d_in[7];
    const float* bn2_b   = (const float*)d_in[8];
    const float* dec2_w  = (const float*)d_in[9];
    const float* bnd2_g  = (const float*)d_in[11];
    const float* bnd2_b  = (const float*)d_in[12];
    const float* dec3_w  = (const float*)d_in[13];
    const float* dec3_b  = (const float*)d_in[14];
    float* out = (float*)d_out;

    // workspace carve-up
    char* ws = (char*)d_ws;
    size_t off = 0;
    auto alloc = [&](size_t bytes) { size_t o = off; off = (off + bytes + 255) & ~(size_t)255; return o; };
    float*    x1     = (float*)(ws + alloc((size_t)BATCH * 128 * L1 * 4));
    float*    x2     = (float*)(ws + alloc((size_t)BATCH * 64 * L2 * 4));
    float*    x3     = (float*)(ws + alloc((size_t)NPOS * 64 * 4));
    float*    ct     = (float*)(ws + alloc((size_t)NPOS * 128 * 4));
    float*    stats  = (float*)(ws + alloc(512 * 4));
    _Float16* hn_h   = (_Float16*)(ws + alloc((size_t)NPOS * 64 * 2));
    _Float16* xz_h   = (_Float16*)(ws + alloc((size_t)NPOS * 512 * 2));
    _Float16* u_h    = (_Float16*)(ws + alloc((size_t)NPOS * 256 * 2));
    float*    x_dbl  = (float*)(ws + alloc((size_t)NPOS * 32 * 4));
    float*    dtbuf  = (float*)(ws + alloc((size_t)NPOS * 256 * 4));   // reused as d2raw
    _Float16* y_h    = (_Float16*)(ws + alloc((size_t)NPOS * 256 * 2));
    float*    tmp64  = (float*)(ws + alloc((size_t)NPOS * 64 * 4));
    float*    d2     = (float*)(ws + alloc((size_t)BATCH * 128 * L1 * 4));
    _Float16* bp_enc2  = (_Float16*)(ws + alloc((size_t)128 * 4 * 512 * 2));
    _Float16* bp_dec2e = (_Float16*)(ws + alloc((size_t)96 * 8 * 512 * 2));
    _Float16* bp_dec2o = (_Float16*)(ws + alloc((size_t)96 * 8 * 512 * 2));
    _Float16* bp_inp[2], *bp_xp[2], *bp_op[2];
    for (int i = 0; i < 2; ++i) {
        bp_inp[i] = (_Float16*)(ws + alloc((size_t)2 * 32 * 512 * 2));
        bp_xp[i]  = (_Float16*)(ws + alloc((size_t)8 * 2 * 512 * 2));
        bp_op[i]  = (_Float16*)(ws + alloc((size_t)8 * 4 * 512 * 2));
    }
    float* d2raw = dtbuf;  // alias: dt is dead once decoder starts

    // ---- pack all GEMM B operands into fragment layout ----
    pack_b_kernel<<<g1(128 * 4 * 512), 256, 0, stream>>>(enc2_w, bp_enc2, 4096, 128, 4, 64, 0, 0);
    pack_b_kernel<<<g1(96 * 8 * 512), 256, 0, stream>>>(dec2_w, bp_dec2e, 3072, 96, 8, 128, 1, 0);
    pack_b_kernel<<<g1(96 * 8 * 512), 256, 0, stream>>>(dec2_w, bp_dec2o, 3072, 96, 8, 128, 1, 1);
    for (int i = 0; i < 2; ++i) {
        const float* inw = (const float*)d_in[15 + 10 * i + 0];
        const float* xpw = (const float*)d_in[15 + 10 * i + 3];
        const float* opw = (const float*)d_in[15 + 10 * i + 8];
        pack_b_kernel<<<g1(2 * 32 * 512), 256, 0, stream>>>(inw, bp_inp[i], 64, 2, 32, 512, 0, 0);
        pack_b_kernel<<<g1(8 * 2 * 512), 256, 0, stream>>>(xpw, bp_xp[i], 256, 8, 2, 20, 0, 0);
        pack_b_kernel<<<g1(8 * 4 * 512), 256, 0, stream>>>(opw, bp_op[i], 256, 8, 4, 64, 0, 0);
    }

    // ---- encoder ----
    enc1_kernel<<<g1((size_t)BATCH * 128 * L1), 256, 0, stream>>>(x, enc1_w, x1);
    bn_stats_kernel<<<128, 256, 0, stream>>>(x1, stats, 128, L1);
    bn_relu_kernel<<<g1((size_t)BATCH * 128 * L1), 256, 0, stream>>>(x1, x1, stats, bn1_g, bn1_b, 128, L1);

    enc2_wmma_kernel<<<dim3(NPOS / 16), 32, 0, stream>>>(x1, bp_enc2, x2);
    bn_stats_kernel<<<64, 256, 0, stream>>>(x2, stats, 64, L2);
    bn_relu_kernel<<<g1((size_t)BATCH * 64 * L2), 256, 0, stream>>>(x2, x2, stats, bn2_g, bn2_b, 64, L2);

    transpose_kernel<<<g1((size_t)NPOS * 64), 256, 0, stream>>>(x2, x3);

    // ---- forward + backward Mamba ----
    for (int i = 0; i < 2; ++i) {
        int dir   = (i == 0) ? 1 : -1;
        int choff = (i == 0) ? 0 : 64;
        const float* conv_w = (const float*)d_in[15 + 10 * i + 1];
        const float* conv_b = (const float*)d_in[15 + 10 * i + 2];
        const float* dtw    = (const float*)d_in[15 + 10 * i + 4];
        const float* dtb    = (const float*)d_in[15 + 10 * i + 5];
        const float* A_log  = (const float*)d_in[15 + 10 * i + 6];
        const float* Dp     = (const float*)d_in[15 + 10 * i + 7];
        const float* nw     = (const float*)d_in[15 + 10 * i + 9];

        rmsnorm_kernel<<<g1(NPOS), 256, 0, stream>>>(x3, nw, hn_h);
        // in_proj: (NPOS x 64) @ (64 x 512) -> xz_h   (32 N-tiles, 4 per wave)
        gemm_wmma_kernel<4><<<dim3(NPOS / 16, 8), 32, 0, stream>>>(hn_h, 64, bp_inp[i], nullptr, xz_h, 512, 2);
        conv_silu_kernel<<<g1((size_t)NPOS * 256), 256, 0, stream>>>(xz_h, conv_w, conv_b, u_h, dir);
        // x_proj: (NPOS x 256) @ (256 x 20pad32) -> x_dbl   (2 N-tiles, 2 per wave)
        gemm_wmma_kernel<2><<<dim3(NPOS / 16, 1), 32, 0, stream>>>(u_h, 256, bp_xp[i], x_dbl, nullptr, 32, 8);
        dt_kernel<<<g1((size_t)NPOS * 256), 256, 0, stream>>>(x_dbl, dtw, dtb, dtbuf);
        scan_kernel<<<BATCH, 256, 0, stream>>>(u_h, dtbuf, x_dbl, xz_h, A_log, Dp, y_h, dir);
        // out_proj: (NPOS x 256) @ (256 x 64) -> tmp64   (4 N-tiles, 4 per wave)
        gemm_wmma_kernel<4><<<dim3(NPOS / 16, 1), 32, 0, stream>>>(y_h, 256, bp_op[i], tmp64, nullptr, 64, 8);
        add_ct_kernel<<<g1((size_t)NPOS * 64), 256, 0, stream>>>(tmp64, x3, ct, choff);
    }

    // ---- decoder ----
    dec2_wmma_kernel<<<dim3(NPOS / 16, 1, 2), 32, 0, stream>>>(ct, x2, bp_dec2e, bp_dec2o, d2raw);
    bn_stats_kernel<<<128, 256, 0, stream>>>(d2raw, stats, 128, L1);
    bn_relu_kernel<<<g1((size_t)BATCH * 128 * L1), 256, 0, stream>>>(d2raw, d2, stats, bnd2_g, bnd2_b, 128, L1);
    dec3_kernel<<<g1((size_t)BATCH * LFULL), 256, 0, stream>>>(d2, x1, dec3_w, dec3_b, out);
}
